// Teacher_S_84945863180991
// MI455X (gfx1250) — matrix operand order repaired and checked by reading, not compiled, not run on hardware
//
#include <hip/hip_runtime.h>
#include <hip/hip_bf16.h>
#include <stdint.h>

typedef __attribute__((ext_vector_type(16))) __bf16       v16bf;
typedef __attribute__((ext_vector_type(8)))  float        v8f;
typedef __attribute__((ext_vector_type(8)))  unsigned int v8u;

union FragBF { v8u u; v16bf b; };
union U2BF   { uint2 u; __bf16 h[4]; };

#define NROW 4096
#define ASTR 72   // LDS row stride (elems) for A tile 128x64
#define BSTR 72   // LDS row stride (elems) for B tile stored transposed [n][k]

// ---------------------------------------------------------------------------
// Edge MLP: softmax([l0,l1])[1] == sigmoid(l1 - l0)
// ---------------------------------------------------------------------------
__device__ __forceinline__ float edge_val(float a, float x, float y, int diag,
    const float* __restrict__ sw1, const float* __restrict__ sb1,
    const float* __restrict__ sw2, const float* __restrict__ sb2) {
  float l0 = sb2[0], l1 = sb2[1];
#pragma unroll
  for (int w = 0; w < 16; ++w) {
    float h = fmaf(a, sw1[w], fmaf(x, sw1[16 + w], fmaf(y, sw1[32 + w], sb1[w])));
    h = fmaxf(h, 0.0f);
    l0 = fmaf(h, sw2[2 * w], l0);
    l1 = fmaf(h, sw2[2 * w + 1], l1);
  }
  float m = 1.0f / (1.0f + __expf(l0 - l1));
  return fmaf(a, m, diag ? 1.0f : 0.0f);
}

__global__ __launch_bounds__(256) void k_edge(
    const float* __restrict__ adj, const float* __restrict__ xd,
    const float* __restrict__ yd,
    const float* __restrict__ w1, const float* __restrict__ b1,
    const float* __restrict__ w2, const float* __restrict__ b2,
    float* __restrict__ af32) {
  __shared__ float sw1[48], sb1[16], sw2[32], sb2[2];
  int tid = threadIdx.x;
  if (tid < 48) sw1[tid] = w1[tid];
  if (tid < 16) sb1[tid] = b1[tid];
  if (tid < 32) sw2[tid] = w2[tid];
  if (tid < 2)  sb2[tid] = b2[tid];
  __syncthreads();
  int i = blockIdx.x;
  size_t base = (size_t)i * NROW;
  for (int j = tid; j < NROW; j += 256)
    af32[base + j] = edge_val(adj[base + j], xd[base + j], yd[base + j],
                              i == j, sw1, sb1, sw2, sb2);
}

__global__ __launch_bounds__(256) void k_rowsum(const float* __restrict__ af32,
                                                float* __restrict__ rowsum) {
  __shared__ float red[256];
  int i = blockIdx.x, tid = threadIdx.x;
  size_t base = (size_t)i * NROW;
  float s = 0.0f;
  for (int j = tid; j < NROW; j += 256) s += af32[base + j];
  red[tid] = s;
  __syncthreads();
  for (int k = 128; k > 0; k >>= 1) {
    if (tid < k) red[tid] += red[tid + k];
    __syncthreads();
  }
  if (tid == 0) rowsum[i] = red[0];
}

// Deterministic two-pass column sum (no float atomics).
__global__ __launch_bounds__(256) void k_colsum_part(
    const float* __restrict__ af32, float* __restrict__ part) {
  int j = blockIdx.x * 256 + threadIdx.x;
  int i0 = blockIdx.y * 128;
  float s = 0.0f;
  for (int i = 0; i < 128; ++i) s += af32[(size_t)(i0 + i) * NROW + j];
  part[(size_t)blockIdx.y * NROW + j] = s;
}

__global__ __launch_bounds__(256) void k_colsum_fin(
    const float* __restrict__ part, float* __restrict__ colsum) {
  int j = blockIdx.x * 256 + threadIdx.x;
  float s = 0.0f;
#pragma unroll
  for (int p = 0; p < 32; ++p) s += part[(size_t)p * NROW + j];
  colsum[j] = s;
}

__global__ __launch_bounds__(256) void k_norm(
    const float* __restrict__ af32, const float* __restrict__ rowsum,
    const float* __restrict__ colsum, __bf16* __restrict__ abf) {
  int i = blockIdx.x, tid = threadIdx.x;
  float rs = rowsum[i];
  float dr = rs > 0.0f ? rsqrtf(rs) : 0.0f;
  size_t base = (size_t)i * NROW;
  for (int j = tid; j < NROW; j += 256) {
    float cs = colsum[j];
    float dc = cs > 0.0f ? rsqrtf(cs) : 0.0f;
    abf[base + j] = (__bf16)(af32[base + j] * dr * dc);
  }
}

__global__ __launch_bounds__(256) void k_pe(const float* __restrict__ w,
                                            const float* __restrict__ b,
                                            __bf16* __restrict__ out,
                                            int cols, size_t n) {
  for (size_t idx = (size_t)blockIdx.x * 256 + threadIdx.x; idx < n;
       idx += (size_t)gridDim.x * 256)
    out[idx] = (__bf16)(w[idx] + b[idx & (size_t)(cols - 1)]);
}

__global__ __launch_bounds__(256) void k_cvt(const float* __restrict__ in,
                                             __bf16* __restrict__ out, size_t n) {
  for (size_t idx = (size_t)blockIdx.x * 256 + threadIdx.x; idx < n;
       idx += (size_t)gridDim.x * 256)
    out[idx] = (__bf16)in[idx];
}

// ---------------------------------------------------------------------------
// bf16 WMMA GEMM: C[M x N] = A[M x K] (row major) @ B[K x N] (row major)
// Block: 256 threads = 8 wave32, tile 128(M) x 64(N), K-tile 64.
// Double-buffered LDS + register prefetch: global loads for tile kt+1 are
// issued before the 8 WMMAs of tile kt; one barrier per K-tile.
// EPI 0: store bf16        EPI 1: BN -> bf16
// EPI 2: +bias -> f32 (mid) and relu -> bf16      EPI 3: BN -> f32
// ---------------------------------------------------------------------------
template <int EPI>
__global__ __launch_bounds__(256) void gemm_bf16(
    const __bf16* __restrict__ A, const __bf16* __restrict__ B,
    int K, int lda, int ldb,
    __bf16* __restrict__ outb, int ldob,
    float* __restrict__ outf, int ldof,
    const float* __restrict__ p0, const float* __restrict__ p1,
    const float* __restrict__ p2, const float* __restrict__ p3) {
  __shared__ __align__(16) __bf16 As[2][128 * ASTR];
  __shared__ __align__(16) __bf16 Bs[2][64 * BSTR];  // transposed: Bs[b][n][k]

  const int tid = threadIdx.x;
  const int wave = tid >> 5, lane = tid & 31;
  const int mr = lane & 15, hi = lane >> 4;
  const int m0b = blockIdx.y * 128;
  const int n0b = blockIdx.x * 64;

  uint2 aR[8];  // A tile 128x64: 2048 uint2, 8 per thread
  uint2 bR[4];  // B tile 64x64 : 1024 uint2, 4 per thread

  auto load_tile = [&](int k0) {
#pragma unroll
    for (int it = 0; it < 8; ++it) {
      int idx = tid + 256 * it;
      int row = idx >> 4, c = (idx & 15) << 2;
      aR[it] = *(const uint2*)&A[(size_t)(m0b + row) * lda + k0 + c];
    }
#pragma unroll
    for (int it = 0; it < 4; ++it) {
      int idx = tid + 256 * it;
      int kr = idx >> 4, c = (idx & 15) << 2;
      bR[it] = *(const uint2*)&B[(size_t)(k0 + kr) * ldb + n0b + c];
    }
  };
  auto store_tile = [&](int buf) {
#pragma unroll
    for (int it = 0; it < 8; ++it) {
      int idx = tid + 256 * it;
      int row = idx >> 4, c = (idx & 15) << 2;
      *(uint2*)&As[buf][row * ASTR + c] = aR[it];
    }
#pragma unroll
    for (int it = 0; it < 4; ++it) {
      int idx = tid + 256 * it;
      int kr = idx >> 4, c = (idx & 15) << 2;
      U2BF tmp; tmp.u = bR[it];
#pragma unroll
      for (int q = 0; q < 4; ++q) Bs[buf][(c + q) * BSTR + kr] = tmp.h[q];
    }
  };

  v8f acc[4] = {v8f{}, v8f{}, v8f{}, v8f{}};

  const int KT = K >> 6;
  load_tile(0);
  store_tile(0);
  __syncthreads();

  for (int kt = 0; kt < KT; ++kt) {
    const int cur = kt & 1;
    if (kt + 1 < KT) load_tile((kt + 1) << 6);  // prefetch to registers

#pragma unroll
    for (int ks = 0; ks < 2; ++ks) {
      // A fragment (ISA 16-bit A 16x32 layout): row m = lane%16; K split by
      // lane half; VGPR v -> K = 2v (v<4) / 16+2(v-4) (v>=4), +8 for lanes>=16.
      FragBF fa;
#pragma unroll
      for (int v = 0; v < 8; ++v) {
        int kk = ks * 32 + (v < 4 ? 2 * v : 16 + 2 * (v - 4)) + (hi ? 8 : 0);
        fa.u[v] = *(const unsigned int*)&As[cur][(wave * 16 + mr) * ASTR + kk];
      }
#pragma unroll
      for (int t = 0; t < 4; ++t) {
        // B fragment: column n = lane%16, VGPR v -> K = 2v (v<4) / 16+2(v-4) (v>=4), +8 for lanes>=16
        FragBF fb;
        int nn = t * 16 + mr;
#pragma unroll
        for (int v = 0; v < 8; ++v) {
          int kk = ks * 32 + (v < 4 ? 2 * v : 16 + 2 * (v - 4)) + (hi ? 8 : 0);
          fb.u[v] = *(const unsigned int*)&Bs[cur][nn * BSTR + kk];
        }
        acc[t] = __builtin_amdgcn_wmma_f32_16x16x32_bf16(
            false, fa.b, false, fb.b, (short)0, acc[t], false, false);
      }
    }

    if (kt + 1 < KT) store_tile(1 - cur);
    __syncthreads();
  }

  // Epilogue: C/D layout — lane%16 = N, reg r -> M = r + 8*(lane>=16)
#pragma unroll
  for (int t = 0; t < 4; ++t) {
    int col = n0b + t * 16 + mr;
    float sc = 1.0f, sh = 0.0f;
    if (EPI == 1 || EPI == 3) {
      sc = p0[col] * rsqrtf(p3[col] + 1e-5f);
      sh = p1[col] - p2[col] * sc;
    } else if (EPI == 2) {
      sh = p0[col];
    }
#pragma unroll
    for (int r = 0; r < 8; ++r) {
      int row = m0b + wave * 16 + hi * 8 + r;
      float v = acc[t][r];
      if (EPI == 0) {
        outb[(size_t)row * ldob + col] = (__bf16)v;
      } else if (EPI == 1) {
        outb[(size_t)row * ldob + col] = (__bf16)(v * sc + sh);
      } else if (EPI == 2) {
        float y = v + sh;
        outf[(size_t)row * ldof + col] = y;
        outb[(size_t)row * ldob + col] = (__bf16)fmaxf(y, 0.0f);
      } else {
        outf[(size_t)row * ldof + col] = v * sc + sh;
      }
    }
  }
}

// ---------------------------------------------------------------------------
extern "C" void kernel_launch(void* const* d_in, const int* in_sizes, int n_in,
                              void* d_out, int out_size, void* d_ws,
                              size_t ws_size, hipStream_t stream) {
  const float* adj    = (const float*)d_in[0];
  const float* xdeg   = (const float*)d_in[1];
  const float* ydeg   = (const float*)d_in[2];
  const float* mlp_w1 = (const float*)d_in[3];
  const float* mlp_b1 = (const float*)d_in[4];
  const float* mlp_w2 = (const float*)d_in[5];
  const float* mlp_b2 = (const float*)d_in[6];
  const float* pe_w   = (const float*)d_in[7];
  const float* pe_b   = (const float*)d_in[8];
  const float* gc1_w  = (const float*)d_in[9];
  const float* lin2_w = (const float*)d_in[10];
  const float* lin2_b = (const float*)d_in[11];
  const float* gc3_w  = (const float*)d_in[12];
  const float* bn1_g  = (const float*)d_in[13];
  const float* bn1_b  = (const float*)d_in[14];
  const float* bn1_m  = (const float*)d_in[15];
  const float* bn1_v  = (const float*)d_in[16];
  const float* bn2_g  = (const float*)d_in[17];
  const float* bn2_b  = (const float*)d_in[18];
  const float* bn2_m  = (const float*)d_in[19];
  const float* bn2_v  = (const float*)d_in[20];

  float* out = (float*)d_out;                    // [4096 x 128]
  float* mid = out + (size_t)4096 * 128;         // [4096 x 64]

  char* w = (char*)d_ws;
  auto carve = [&](size_t bytes) -> char* {
    char* p = w;
    w += (bytes + 255) & ~(size_t)255;
    return p;
  };
  float*  af32   = (float*) carve((size_t)4096 * 4096 * 4);
  __bf16* abf    = (__bf16*)carve((size_t)4096 * 4096 * 2);
  __bf16* pebf   = (__bf16*)carve((size_t)4096 * 512 * 2);
  __bf16* gc1bf  = (__bf16*)carve((size_t)512 * 512 * 2);
  __bf16* b1bf   = (__bf16*)carve((size_t)4096 * 512 * 2);
  __bf16* h1bf   = (__bf16*)carve((size_t)4096 * 512 * 2);
  __bf16* lin2bf = (__bf16*)carve((size_t)512 * 64 * 2);
  __bf16* rbf    = (__bf16*)carve((size_t)4096 * 64 * 2);
  __bf16* gc3bf  = (__bf16*)carve((size_t)64 * 128 * 2);
  __bf16* c3bf   = (__bf16*)carve((size_t)4096 * 128 * 2);
  float*  rowsum = (float*) carve((size_t)4096 * 4);
  float*  colsum = (float*) carve((size_t)4096 * 4);
  float*  colprt = (float*) carve((size_t)32 * 4096 * 4);

  // a = D^{-1/2} (adj*mask + I) D^{-1/2}  -> bf16
  k_edge<<<4096, 256, 0, stream>>>(adj, xdeg, ydeg, mlp_w1, mlp_b1, mlp_w2,
                                   mlp_b2, af32);
  k_rowsum<<<4096, 256, 0, stream>>>(af32, rowsum);
  k_colsum_part<<<dim3(16, 32), 256, 0, stream>>>(af32, colprt);
  k_colsum_fin<<<16, 256, 0, stream>>>(colprt, colsum);
  k_norm<<<4096, 256, 0, stream>>>(af32, rowsum, colsum, abf);

  // operand conversion fp32 -> bf16
  k_pe<<<2048, 256, 0, stream>>>(pe_w, pe_b, pebf, 512, (size_t)4096 * 512);
  k_cvt<<<256, 256, 0, stream>>>(gc1_w, gc1bf, (size_t)512 * 512);
  k_cvt<<<32, 256, 0, stream>>>(lin2_w, lin2bf, (size_t)512 * 64);
  k_cvt<<<8, 256, 0, stream>>>(gc3_w, gc3bf, (size_t)64 * 128);

  // G1: B1 = pe @ gc1            [4096x512] = [4096x512]@[512x512]
  gemm_bf16<0><<<dim3(8, 32), 256, 0, stream>>>(
      pebf, gc1bf, 512, 512, 512, b1bf, 512, nullptr, 0,
      nullptr, nullptr, nullptr, nullptr);
  // G2: H1 = BN1(a @ B1)         [4096x512], K=4096
  gemm_bf16<1><<<dim3(8, 32), 256, 0, stream>>>(
      abf, b1bf, 4096, 4096, 512, h1bf, 512, nullptr, 0,
      bn1_g, bn1_b, bn1_m, bn1_v);
  // G3: mid = H1 @ lin2 + b (f32 out) ; R = relu(mid) bf16   [4096x64]
  gemm_bf16<2><<<dim3(1, 32), 256, 0, stream>>>(
      h1bf, lin2bf, 512, 512, 64, rbf, 64, mid, 64,
      lin2_b, nullptr, nullptr, nullptr);
  // G4: C3 = R @ gc3             [4096x128], K=64
  gemm_bf16<0><<<dim3(2, 32), 256, 0, stream>>>(
      rbf, gc3bf, 64, 64, 128, c3bf, 128, nullptr, 0,
      nullptr, nullptr, nullptr, nullptr);
  // G5: out = BN2(a @ C3)        [4096x128], K=4096
  gemm_bf16<3><<<dim3(2, 32), 256, 0, stream>>>(
      abf, c3bf, 4096, 4096, 128, nullptr, 0, out, 128,
      bn2_g, bn2_b, bn2_m, bn2_v);
}
